// GroupedQueryAttention_13116830122562
// MI455X (gfx1250) — matrix operand order repaired and checked by reading, not compiled, not run on hardware
//
#include <hip/hip_runtime.h>

// ---------------------------------------------------------------------------
// Problem constants (reference: GQA, B=2, S=2048, E=2048, H=16, G=4, D=128)
// ---------------------------------------------------------------------------
#define BDIM   2
#define SDIM   2048
#define EDIM   2048
#define HHEADS 16
#define GGRP   4
#define DHEAD  128
#define KVDIM  (GGRP * DHEAD)          // 512
#define MROWS  (BDIM * SDIM)           // 4096

typedef __attribute__((ext_vector_type(16))) _Float16 v16h;
typedef __attribute__((ext_vector_type(8)))  float    v8f;

union FragU { v16h v; uint4 q[2]; };

// ---------------------------------------------------------------------------
// WMMA fragment helpers (CDNA5 ISA 7.12.2 layouts, wave32)
// ---------------------------------------------------------------------------
// A-matrix 16x32 f16: lanes 0-15 hold row M=lane with K=[0..7] in dwords 0-3
// and K=[16..23] in dwords 4-7; lanes 16-31 hold K=[8..15] and K=[24..31].
__device__ __forceinline__ v16h load_frag_a(const _Float16* base, int ld) {
  const int lane = threadIdx.x & 31;
  const int row  = lane & 15;
  const int ko   = (lane >> 4) << 3;                 // 0 or 8
  const _Float16* p = base + (size_t)row * ld + ko;
  FragU u;
  u.q[0] = *reinterpret_cast<const uint4*>(p);       // K ko..ko+7
  u.q[1] = *reinterpret_cast<const uint4*>(p + 16);  // K ko+16..ko+23
  return u.v;
}

// B-matrix 32x16 f16 given as B^T rows (row n = column n of B, K contiguous):
// lanes 0-15 hold column N=lane with K=[0..7],[16..23]; lanes 16-31 hold K=[8..15],[24..31].
__device__ __forceinline__ v16h load_frag_b(const _Float16* baseT, int ld) {
  const int lane = threadIdx.x & 31;
  const int col  = lane & 15;
  const int ko   = (lane >> 4) << 3;                 // 0 or 8
  const _Float16* p = baseT + (size_t)col * ld + ko;
  FragU u;
  u.q[0] = *reinterpret_cast<const uint4*>(p);       // K ko..ko+7
  u.q[1] = *reinterpret_cast<const uint4*>(p + 16);  // K ko+16..ko+23
  return u.v;
}

__device__ __forceinline__ v8f wmma16(v16h a, v16h b, v8f c) {
  // D = A(16x32) * B(32x16) + C, f32 accumulate
  return __builtin_amdgcn_wmma_f32_16x16x32_f16(false, a, false, b,
                                                (short)0, c, false, false);
}

// Async copy 16B global -> LDS (GLOBAL_LOAD_ASYNC_TO_LDS_B128, ASYNCcnt).
// VDST = per-lane LDS byte address (low 32 bits of the generic pointer are
// the LDS offset per the aperture mapping), VADDR = 64-bit global address.
__device__ __forceinline__ void async_copy16(const _Float16* g, _Float16* l) {
  uint32_t lo = (uint32_t)(uintptr_t)l;
  uint64_t ga = (uint64_t)(uintptr_t)g;
  asm volatile("global_load_async_to_lds_b128 %0, %1, off"
               :: "v"(lo), "v"(ga) : "memory");
}

// ---------------------------------------------------------------------------
// Conversion / transpose kernels
// ---------------------------------------------------------------------------
__global__ void cvt_f32_f16(const float* __restrict__ in,
                            _Float16* __restrict__ out, long n) {
  long i = (long)blockIdx.x * blockDim.x + threadIdx.x;
  long stride = (long)gridDim.x * blockDim.x;
  for (; i < n; i += stride) out[i] = (_Float16)in[i];
}

// W: [K,N] f32 row-major  ->  Wt: [N,K] f16 row-major
__global__ void transpose_cvt(const float* __restrict__ W,
                              _Float16* __restrict__ Wt, int K, int N) {
  long total = (long)K * N;
  long i = (long)blockIdx.x * blockDim.x + threadIdx.x;
  long stride = (long)gridDim.x * blockDim.x;
  for (; i < total; i += stride) {
    int k = (int)(i / N);
    int n = (int)(i % N);
    Wt[(long)n * K + k] = (_Float16)W[i];
  }
}

// V: [B*S, KV] f16  ->  Vt: [B,G,D,S] f16
__global__ void transpose_v(const _Float16* __restrict__ V,
                            _Float16* __restrict__ Vt) {
  long total = (long)MROWS * KVDIM;
  long i = (long)blockIdx.x * blockDim.x + threadIdx.x;
  long stride = (long)gridDim.x * blockDim.x;
  for (; i < total; i += stride) {
    int row = (int)(i / KVDIM);            // b*S + s
    int c   = (int)(i % KVDIM);            // g*D + d
    int b = row / SDIM, s = row % SDIM;
    int g = c / DHEAD,  d = c % DHEAD;
    Vt[(((long)(b * GGRP + g)) * DHEAD + d) * SDIM + s] = V[i];
  }
}

// ---------------------------------------------------------------------------
// Tiled WMMA GEMM with async-LDS double buffering:
//   C[M,N] = A[M,K] (f16) @ Bt[N,K]^T (f16) + bias
// block = 256 threads = 8 waves; block tile 128x128; wave tile 32x64 (2x4).
// Per k-step (32): A/B tiles (8KB each) staged via global_load_async_to_lds,
// double buffered: issue tile k+1, s_wait_asynccnt for tile k, barrier, WMMA.
// mode 0: f32 output   mode 1: f16 output
// ---------------------------------------------------------------------------
__global__ __launch_bounds__(256) void gemm_wmma_f16(
    const _Float16* __restrict__ A, const _Float16* __restrict__ Bt,
    const float* __restrict__ bias, void* __restrict__ out,
    int M, int N, int K, int mode) {
  __shared__ _Float16 sA[2][128 * 32];   // 16 KB
  __shared__ _Float16 sB[2][128 * 32];   // 16 KB

  const int tid  = threadIdx.x;
  const int wave = tid >> 5;
  const int lane = tid & 31;
  const int hh = lane >> 4;              // half-wave: 0 or 1
  const int ln = lane & 15;
  const int wm = wave >> 1;              // 0..3 -> 32-row strip
  const int wn = wave & 1;               // 0..1 -> 64-col strip

  const _Float16* Ablk = A  + (size_t)(blockIdx.y * 128) * K;
  const _Float16* Bblk = Bt + (size_t)(blockIdx.x * 128) * K;

  // Each thread copies 2 x 16B chunks of A and 2 of B per k-step
  // (4 async instructions per wave per tile).
  const int r0 = tid >> 2,        s0c = tid & 3;          // chunk tid
  const int r1 = (tid + 256) >> 2, s1c = (tid + 256) & 3; // chunk tid+256

#define ISSUE_TILE(kt, buf)                                                  \
  do {                                                                       \
    int kc = (kt) * 32;                                                      \
    async_copy16(Ablk + (size_t)r0 * K + kc + s0c * 8,                       \
                 &sA[buf][r0 * 32 + s0c * 8]);                               \
    async_copy16(Ablk + (size_t)r1 * K + kc + s1c * 8,                       \
                 &sA[buf][r1 * 32 + s1c * 8]);                               \
    async_copy16(Bblk + (size_t)r0 * K + kc + s0c * 8,                       \
                 &sB[buf][r0 * 32 + s0c * 8]);                               \
    async_copy16(Bblk + (size_t)r1 * K + kc + s1c * 8,                       \
                 &sB[buf][r1 * 32 + s1c * 8]);                               \
  } while (0)

  v8f z = {0.f, 0.f, 0.f, 0.f, 0.f, 0.f, 0.f, 0.f};
  v8f acc[2][4];
#pragma unroll
  for (int i = 0; i < 2; ++i)
#pragma unroll
    for (int j = 0; j < 4; ++j) acc[i][j] = z;

  const int nk = K >> 5;
  ISSUE_TILE(0, 0);

  for (int kt = 0; kt < nk; ++kt) {
    const int cur = kt & 1;
    if (kt + 1 < nk) {
      ISSUE_TILE(kt + 1, cur ^ 1);
      asm volatile("s_wait_asynccnt 4" ::: "memory");  // tile kt done (in-order)
    } else {
      asm volatile("s_wait_asynccnt 0" ::: "memory");
    }
    __syncthreads();

    const _Float16* sAc = &sA[cur][0];
    const _Float16* sBc = &sB[cur][0];
    v16h a0 = load_frag_a(sAc + (wm * 32) * 32, 32);
    v16h a1 = load_frag_a(sAc + (wm * 32 + 16) * 32, 32);
#pragma unroll
    for (int j = 0; j < 4; ++j) {
      v16h b = load_frag_b(sBc + (wn * 64 + 16 * j) * 32, 32);
      acc[0][j] = wmma16(a0, b, acc[0][j]);
      acc[1][j] = wmma16(a1, b, acc[1][j]);
    }
    __syncthreads();   // all reads of `cur` done before it is refilled
  }
#undef ISSUE_TILE

  const int m0 = blockIdx.y * 128 + wm * 32;
  const int n0 = blockIdx.x * 128 + wn * 64;
#pragma unroll
  for (int i = 0; i < 2; ++i)
#pragma unroll
    for (int j = 0; j < 4; ++j)
#pragma unroll
      for (int r = 0; r < 8; ++r) {
        int m = m0 + 16 * i + r + 8 * hh;    // C-layout: VGPR r, halves
        int n = n0 + 16 * j + ln;
        float v = acc[i][j][r] + (bias ? bias[n] : 0.f);
        if (mode == 0) ((float*)out)[(size_t)m * N + n] = v;
        else           ((_Float16*)out)[(size_t)m * N + n] = (_Float16)v;
      }
}

// ---------------------------------------------------------------------------
// Flash attention (per wave: 16 query rows, full online softmax)
// grid = (S/64, B*H), block = 128 (4 independent waves)
// Q:[B,S,E] f16   K:[B,S,KV] f16   Vt:[B,G,D,S] f16   O:[B,S,E] f16
// ---------------------------------------------------------------------------
__global__ __launch_bounds__(128) void fa_wmma(
    const _Float16* __restrict__ Q, const _Float16* __restrict__ Km,
    const _Float16* __restrict__ Vt, _Float16* __restrict__ O) {
  __shared__ _Float16 lds[4][16 * 64];     // per-wave P staging (2 KB each)

  const int wave = threadIdx.x >> 5;
  const int lane = threadIdx.x & 31;
  const int hh = lane >> 4;
  const int ln = lane & 15;
  const int bh = blockIdx.y;
  const int b = bh / HHEADS, h = bh % HHEADS;
  const int g = h >> 2;                    // R = H/G = 4 heads per kv group
  const int q0 = blockIdx.x * 64 + wave * 16;
  const float scale = 0.08838834764831845f;  // 1/sqrt(128)

  // Resident Q fragments: 16 rows x D=128 -> 4 K-chunks
  v16h qf[4];
  const _Float16* qbase = Q + ((size_t)(b * SDIM + q0)) * EDIM + h * DHEAD;
#pragma unroll
  for (int kk = 0; kk < 4; ++kk) qf[kk] = load_frag_a(qbase + kk * 32, EDIM);

  v8f z = {0.f, 0.f, 0.f, 0.f, 0.f, 0.f, 0.f, 0.f};
  v8f o[8];
#pragma unroll
  for (int j = 0; j < 8; ++j) o[j] = z;
  float mr[8], lr[8];
#pragma unroll
  for (int r = 0; r < 8; ++r) { mr[r] = -1e30f; lr[r] = 0.f; }

  _Float16* myl = lds[wave];
  const _Float16* vbase =
      Vt + ((size_t)(b * GGRP + g)) * DHEAD * (size_t)SDIM;

  for (int t0 = 0; t0 < SDIM; t0 += 64) {
    // ---- S = Q K^T for a 16x64 score tile (16 WMMAs) ----
    v8f sc[4];
#pragma unroll
    for (int j = 0; j < 4; ++j) sc[j] = z;
    const _Float16* kbase = Km + ((size_t)(b * SDIM + t0)) * KVDIM + g * DHEAD;
#pragma unroll
    for (int kk = 0; kk < 4; ++kk)
#pragma unroll
      for (int j = 0; j < 4; ++j) {
        v16h bf = load_frag_b(kbase + (size_t)(16 * j) * KVDIM + kk * 32, KVDIM);
        sc[j] = wmma16(qf[kk], bf, sc[j]);
      }

    // ---- online softmax in C-fragment layout ----
    float corr[8];
#pragma unroll
    for (int r = 0; r < 8; ++r) {
      float vmax = -1e30f;
#pragma unroll
      for (int j = 0; j < 4; ++j) vmax = fmaxf(vmax, sc[j][r] * scale);
#pragma unroll
      for (int msk = 1; msk < 16; msk <<= 1)
        vmax = fmaxf(vmax, __shfl_xor(vmax, msk, 32));
      float nm = fmaxf(mr[r], vmax);
      corr[r] = __expf(mr[r] - nm);
      mr[r] = nm;
    }
    float rs[8];
#pragma unroll
    for (int r = 0; r < 8; ++r) rs[r] = 0.f;
#pragma unroll
    for (int j = 0; j < 4; ++j)
#pragma unroll
      for (int r = 0; r < 8; ++r) {
        float p = __expf(sc[j][r] * scale - mr[r]);
        sc[j][r] = p;
        rs[r] += p;
      }
#pragma unroll
    for (int r = 0; r < 8; ++r) {
#pragma unroll
      for (int msk = 1; msk < 16; msk <<= 1) rs[r] += __shfl_xor(rs[r], msk, 32);
      lr[r] = lr[r] * corr[r] + rs[r];
#pragma unroll
      for (int j = 0; j < 8; ++j) o[j][r] *= corr[r];
    }

    // ---- stage P (f16) through per-wave LDS: C-layout -> A-layout ----
#pragma unroll
    for (int j = 0; j < 4; ++j)
#pragma unroll
      for (int r = 0; r < 8; ++r)
        myl[(r + 8 * hh) * 64 + 16 * j + ln] = (_Float16)sc[j][r];
    asm volatile("s_wait_dscnt 0" ::: "memory");
    v16h pf[2];
    pf[0] = load_frag_a(myl, 64);
    pf[1] = load_frag_a(myl + 32, 64);
    asm volatile("s_wait_dscnt 0" ::: "memory");

    // ---- O += P V  (16 WMMAs: 8 d-frags x 2 key-chunks) ----
#pragma unroll
    for (int j2 = 0; j2 < 8; ++j2)
#pragma unroll
      for (int kk2 = 0; kk2 < 2; ++kk2) {
        v16h vf = load_frag_b(vbase + (size_t)(16 * j2) * SDIM + t0 + kk2 * 32,
                              SDIM);
        o[j2] = wmma16(pf[kk2], vf, o[j2]);
      }
  }

  // ---- normalize and store [B,S,E] f16 ----
  float inv[8];
#pragma unroll
  for (int r = 0; r < 8; ++r) inv[r] = 1.f / lr[r];
#pragma unroll
  for (int j2 = 0; j2 < 8; ++j2)
#pragma unroll
    for (int r = 0; r < 8; ++r) {
      int sq = q0 + r + 8 * hh;
      int col = h * DHEAD + 16 * j2 + ln;
      O[((size_t)(b * SDIM + sq)) * EDIM + col] = (_Float16)(o[j2][r] * inv[r]);
    }
}

// ---------------------------------------------------------------------------
// Host-side launch
// ---------------------------------------------------------------------------
extern "C" void kernel_launch(void* const* d_in, const int* in_sizes, int n_in,
                              void* d_out, int out_size, void* d_ws,
                              size_t ws_size, hipStream_t stream) {
  (void)in_sizes; (void)n_in; (void)out_size; (void)ws_size;
  const float* x  = (const float*)d_in[0];
  const float* Wq = (const float*)d_in[1];
  const float* bq = (const float*)d_in[2];
  const float* Wk = (const float*)d_in[3];
  const float* bk = (const float*)d_in[4];
  const float* Wv = (const float*)d_in[5];
  const float* bv = (const float*)d_in[6];
  const float* Wo = (const float*)d_in[7];
  const float* bo = (const float*)d_in[8];
  float* out = (float*)d_out;

  // Workspace layout (f16 elements)
  _Float16* ws = (_Float16*)d_ws;
  const size_t NX  = (size_t)MROWS * EDIM;    // 8M
  const size_t NWQ = (size_t)EDIM * EDIM;     // 4M
  const size_t NWK = (size_t)KVDIM * EDIM;    // 1M
  const size_t NQK = (size_t)MROWS * KVDIM;   // 2M
  _Float16* xb   = ws;                 ws += NX;
  _Float16* Wqt  = ws;                 ws += NWQ;
  _Float16* Wkt  = ws;                 ws += NWK;
  _Float16* Wvt  = ws;                 ws += NWK;
  _Float16* Wot  = ws;                 ws += NWQ;
  _Float16* Qb   = ws;                 ws += NX;
  _Float16* Kb   = ws;                 ws += NQK;
  _Float16* Vb   = ws;                 ws += NQK;
  _Float16* Vtb  = ws;                 ws += NQK;
  _Float16* attn = ws;                 ws += NX;

  // 1) convert x, transpose-convert weights to f16
  cvt_f32_f16<<<2048, 256, 0, stream>>>(x, xb, (long)NX);
  transpose_cvt<<<2048, 256, 0, stream>>>(Wq, Wqt, EDIM, EDIM);
  transpose_cvt<<<1024, 256, 0, stream>>>(Wk, Wkt, EDIM, KVDIM);
  transpose_cvt<<<1024, 256, 0, stream>>>(Wv, Wvt, EDIM, KVDIM);
  transpose_cvt<<<2048, 256, 0, stream>>>(Wo, Wot, EDIM, EDIM);

  // 2) projections (WMMA GEMMs, f16 out), block tile 128x128
  gemm_wmma_f16<<<dim3(EDIM / 128, MROWS / 128), 256, 0, stream>>>(
      xb, Wqt, bq, (void*)Qb, MROWS, EDIM, EDIM, 1);
  gemm_wmma_f16<<<dim3(KVDIM / 128, MROWS / 128), 256, 0, stream>>>(
      xb, Wkt, bk, (void*)Kb, MROWS, KVDIM, EDIM, 1);
  gemm_wmma_f16<<<dim3(KVDIM / 128, MROWS / 128), 256, 0, stream>>>(
      xb, Wvt, bv, (void*)Vb, MROWS, KVDIM, EDIM, 1);

  // 3) V -> V^T per (b,g) head-group for contiguous PV B-fragments
  transpose_v<<<2048, 256, 0, stream>>>(Vb, Vtb);

  // 4) flash attention (WMMA QK^T + online softmax + WMMA PV)
  fa_wmma<<<dim3(SDIM / 64, BDIM * HHEADS), 128, 0, stream>>>(Qb, Kb, Vtb, attn);

  // 5) output projection (WMMA GEMM, f32 out + bias -> d_out)
  gemm_wmma_f16<<<dim3(EDIM / 128, MROWS / 128), 256, 0, stream>>>(
      attn, Wot, bo, (void*)out, MROWS, EDIM, EDIM, 0);
}
